// ChannelAttExchange_7602092114037
// MI455X (gfx1250) — matrix operand order repaired and checked by reading, not compiled, not run on hardware
//
#include <hip/hip_runtime.h>
#include <hip/hip_bf16.h>

// ---------------------------------------------------------------------------
// ChannelAttExchange for MI455X (gfx1250, wave32, WMMA).
// GEMM-shaped ops (1x1 convs, per-pixel MLP) use v_wmma_f32_16x16x32_bf16 with
// pre-packed bf16 A-fragments (b128 loads, no per-step conversion VALU).
// convfull_reduce additionally prefetches its x-tile into LDS with
// global_load_async_to_lds_b128 (ASYNCcnt), overlapped behind the GEMM.
// ---------------------------------------------------------------------------

typedef unsigned short u16;
typedef unsigned int   u32;
typedef __attribute__((ext_vector_type(16))) __bf16 v16bf;
typedef __attribute__((ext_vector_type(8)))  float  v8f;
typedef __attribute__((ext_vector_type(8)))  u32    v8u;
typedef __attribute__((ext_vector_type(4)))  u32    v4u;

#define DEVINL __device__ __forceinline__

constexpr int    kN    = 8;
constexpr int    kC    = 256;
constexpr int    kH    = 128;
constexpr int    kW    = 128;
constexpr int    kHW   = kH * kW;             // 16384
constexpr int    kCH   = 128;                 // C/2 == K (exchange channels)
constexpr int    kHID  = 64;
constexpr size_t kNCHW = (size_t)kN * kC * kHW;
constexpr size_t kNKHW = (size_t)kN * kCH * kHW;
constexpr int    kT16PerImg = kHW / 16;       // 1024  (16-px tiles, MLP)
constexpr int    kT16       = kN * kT16PerImg;
constexpr int    kT32PerImg = kHW / 32;       // 512   (32-px tiles, convs)
constexpr int    kT32       = kN * kT32PerImg;

// ---------------- async global->LDS (gfx1250 ASYNCcnt path) ----------------
#if defined(__AMDGCN__) && __has_builtin(__builtin_amdgcn_global_load_async_to_lds_b128)
#define HAS_ASYNC_LDS 1
typedef int v4ig __attribute__((vector_size(16)));            // V4i (GCC vector)
typedef __attribute__((address_space(1))) v4ig* gv4i_p;       // global V4i*
typedef __attribute__((address_space(3))) v4ig* lv4i_p;       // LDS V4i*
DEVINL void async_cp16(const float* g, float* l) {
  __builtin_amdgcn_global_load_async_to_lds_b128((gv4i_p)g, (lv4i_p)l, 0, 0);
}
DEVINL void wait_async_zero() {
#if __has_builtin(__builtin_amdgcn_s_wait_asynccnt)
  __builtin_amdgcn_s_wait_asynccnt(0);
#else
  asm volatile("s_wait_asynccnt 0" ::: "memory");
#endif
}
#else
#define HAS_ASYNC_LDS 0
#endif

// ---------------- bf16 helpers (round-to-nearest-even pack) ----------------
DEVINL u16 f2bf(float f) {
  u32 u = __builtin_bit_cast(u32, f);
  u32 r = u + 0x7FFFu + ((u >> 16) & 1u);
  return (u16)(r >> 16);
}
DEVINL float bf2f(u16 h) {
  u32 u = ((u32)h) << 16;
  return __builtin_bit_cast(float, u);
}

// ---------------- WMMA fragment plumbing (CDNA5 16x16x32 bf16) -------------
// 16 contiguous bf16 (32B, 16B-aligned) -> v16bf via two b128 loads.
DEVINL v16bf ld_frag16(const u16* __restrict__ p) {
  v4u a = *(const v4u*)p;
  v4u b = *(const v4u*)(p + 8);
  v8u w;
  w[0] = a[0]; w[1] = a[1]; w[2] = a[2]; w[3] = a[3];
  w[4] = b[0]; w[5] = b[1]; w[6] = b[2]; w[7] = b[3];
  return __builtin_bit_cast(v16bf, w);
}
// A fragments pre-packed fragment-major: [frag][lane][16]
DEVINL v16bf load_a_pk(const u16* __restrict__ Apk, int frag, int lane) {
  return ld_frag16(Apk + ((size_t)frag << 9) + (lane << 4));
}
// B (32x16, KxN) from column-major LDS (stride KDP halves, KDP % 8 == 0):
// lane L holds col N = L&15; element e -> k = e + (L>=16 ? 16 : 0).
DEVINL v16bf load_b_lds(const u16* __restrict__ colBase, int k0, int lane) {
  return ld_frag16(colBase + k0 + ((lane >= 16) ? 16 : 0));
}
DEVINL v8f wmma_bf16(v16bf a, v16bf b, v8f c) {
  return __builtin_amdgcn_wmma_f32_16x16x32_bf16(false, a, false, b, (short)0, c,
                                                 false, false);
}

// ---------------------------------------------------------------------------
// Kernel 0: pack f32 weight matrix (M x KD, row-major) into A-fragment order.
// A (16x32): lane L holds row M = L&15, reg elem e:
//   k = e + (L>=16 ? 16 : 0).
// ---------------------------------------------------------------------------
__global__ void pack_a_kernel(const float* __restrict__ Wm, u16* __restrict__ Apk,
                              int M, int KD) {
  const int i = blockIdx.x * 256 + threadIdx.x;   // over M*KD
  if (i >= M * KD) return;
  const int e    = i & 15;
  const int lane = (i >> 4) & 31;
  const int frag = i >> 9;
  const int ksteps = KD >> 5;
  const int mtile = frag / ksteps, kstep = frag - mtile * ksteps;
  const int row = (mtile << 4) + (lane & 15);
  const int k = (kstep << 5) + e + ((lane >= 16) ? 16 : 0);
  Apk[i] = f2bf(Wm[(size_t)row * KD + k]);
}

// ---------------------------------------------------------------------------
// Kernel 1: depthwise 5x5, pad 2.  x f32 -> a1 bf16
// ---------------------------------------------------------------------------
__global__ void dw5_kernel(const float* __restrict__ x, const float* __restrict__ w,
                           const float* __restrict__ b, u16* __restrict__ out) {
  const int nc = blockIdx.y;                    // n*C + c
  const int c  = nc & (kC - 1);
  const int p  = blockIdx.x * 256 + threadIdx.x;
  const int h  = p >> 7, ww = p & 127;
  const float* xp = x + (size_t)nc * kHW;
  const float* wc = w + c * 25;
  __builtin_prefetch(xp + p + 256, 0, 0);       // global_prefetch_b8
  float acc = b[c];
#pragma unroll
  for (int ky = 0; ky < 5; ++ky) {
    int yy = h + ky - 2;
    if ((unsigned)yy < (unsigned)kH) {
#pragma unroll
      for (int kx = 0; kx < 5; ++kx) {
        int xx = ww + kx - 2;
        if ((unsigned)xx < (unsigned)kW)
          acc += wc[ky * 5 + kx] * xp[yy * kW + xx];
      }
    }
  }
  out[(size_t)nc * kHW + p] = f2bf(acc);
}

// ---------------------------------------------------------------------------
// Kernel 2: depthwise 7x7, dilation 3, pad 9.  a1 bf16 -> a2 bf16
// ---------------------------------------------------------------------------
__global__ void dw7d3_kernel(const u16* __restrict__ in, const float* __restrict__ w,
                             const float* __restrict__ b, u16* __restrict__ out) {
  const int nc = blockIdx.y;
  const int c  = nc & (kC - 1);
  const int p  = blockIdx.x * 256 + threadIdx.x;
  const int h  = p >> 7, ww = p & 127;
  const u16* xp = in + (size_t)nc * kHW;
  const float* wc = w + c * 49;
  float acc = b[c];
#pragma unroll
  for (int ky = 0; ky < 7; ++ky) {
    int yy = h + ky * 3 - 9;
    if ((unsigned)yy < (unsigned)kH) {
#pragma unroll
      for (int kx = 0; kx < 7; ++kx) {
        int xx = ww + kx * 3 - 9;
        if ((unsigned)xx < (unsigned)kW)
          acc += wc[ky * 7 + kx] * bf2f(xp[yy * kW + xx]);
      }
    }
  }
  out[(size_t)nc * kHW + p] = f2bf(acc);
}

// ---------------------------------------------------------------------------
// Kernel 3: 1x1 conv 256 -> 128 via WMMA.  32 px/block: each wave owns one
// 16-row M-tile and two 16-px C tiles sharing every A fragment (2 wmma / A).
// ---------------------------------------------------------------------------
__global__ void __launch_bounds__(256)
conv1x1_half_kernel(const u16* __restrict__ in, const u16* __restrict__ Apk,
                    const float* __restrict__ bias, u16* __restrict__ out) {
  constexpr int KD = 256, KDP = KD + 8;         // 528B col stride: 16B skew
  __shared__ u16 Bsh[32 * KDP];
  const int tile = blockIdx.x;
  const int n  = tile >> 9;                     // 512 tiles / image
  const int p0 = (tile & 511) << 5;
  const int tid = threadIdx.x;
  {  // stage: thread t = channel t, 32 px as 4x b128, transpose into LDS
    const u16* src = in + (size_t)(n * kC + tid) * kHW + p0;
    v4u d0 = *(const v4u*)src;
    v4u d1 = *(const v4u*)(src + 8);
    v4u d2 = *(const v4u*)(src + 16);
    v4u d3 = *(const v4u*)(src + 24);
#pragma unroll
    for (int q = 0; q < 4; ++q) {
      Bsh[(2 * q + 0)  * KDP + tid] = (u16)(d0[q] & 0xffffu);
      Bsh[(2 * q + 1)  * KDP + tid] = (u16)(d0[q] >> 16);
      Bsh[(2 * q + 8)  * KDP + tid] = (u16)(d1[q] & 0xffffu);
      Bsh[(2 * q + 9)  * KDP + tid] = (u16)(d1[q] >> 16);
      Bsh[(2 * q + 16) * KDP + tid] = (u16)(d2[q] & 0xffffu);
      Bsh[(2 * q + 17) * KDP + tid] = (u16)(d2[q] >> 16);
      Bsh[(2 * q + 24) * KDP + tid] = (u16)(d3[q] & 0xffffu);
      Bsh[(2 * q + 25) * KDP + tid] = (u16)(d3[q] >> 16);
    }
  }
  __syncthreads();
  const int wave = tid >> 5, lane = tid & 31, col = lane & 15;
  const int m0 = wave << 4;
  const int rbase = m0 + ((lane >= 16) ? 8 : 0);
  v8f acc0, acc1;
#pragma unroll
  for (int r = 0; r < 8; ++r) { float bv = bias[rbase + r]; acc0[r] = bv; acc1[r] = bv; }
  const u16* cb0 = Bsh + col * KDP;
  const u16* cb1 = Bsh + (col + 16) * KDP;
#pragma unroll
  for (int ks = 0; ks < 8; ++ks) {
    v16bf a  = load_a_pk(Apk, wave * 8 + ks, lane);
    acc0 = wmma_bf16(a, load_b_lds(cb0, ks * 32, lane), acc0);
    acc1 = wmma_bf16(a, load_b_lds(cb1, ks * 32, lane), acc1);
  }
  const size_t ob = (size_t)(n * kCH + rbase) * kHW + p0 + col;
#pragma unroll
  for (int r = 0; r < 8; ++r) {
    out[ob + (size_t)r * kHW]      = f2bf(acc0[r]);
    out[ob + (size_t)r * kHW + 16] = f2bf(acc1[r]);
  }
}

// ---------------------------------------------------------------------------
// Kernel 4: channel avg/max over concat(a1c,a2c) per pixel -> agg (N,2,HW)
// ---------------------------------------------------------------------------
__global__ void agg_kernel(const u16* __restrict__ a1c, const u16* __restrict__ a2c,
                           float* __restrict__ agg) {
  const int n = blockIdx.y;
  const int p = blockIdx.x * 256 + threadIdx.x;
  const u16* p1 = a1c + (size_t)n * kCH * kHW + p;
  const u16* p2 = a2c + (size_t)n * kCH * kHW + p;
  float s = 0.f, m = -3.4e38f;
  for (int c = 0; c < kCH; ++c) {
    float v = bf2f(p1[(size_t)c * kHW]); s += v; m = fmaxf(m, v);
  }
  for (int c = 0; c < kCH; ++c) {
    float v = bf2f(p2[(size_t)c * kHW]); s += v; m = fmaxf(m, v);
  }
  agg[(size_t)(n * 2) * kHW + p]     = s * (1.f / 256.f);
  agg[(size_t)(n * 2 + 1) * kHW + p] = m;
}

// ---------------------------------------------------------------------------
// Kernel 5: 7x7 conv (2->2 ch, pad 3) on agg + sigmoid -> sig (N,2,HW)
// ---------------------------------------------------------------------------
__global__ void __launch_bounds__(256)
squeeze_kernel(const float* __restrict__ agg, const float* __restrict__ wsq,
               const float* __restrict__ bsq, float* __restrict__ sig) {
  __shared__ float ws[198];
  const int n = blockIdx.y;
  const int tid = threadIdx.x;
  if (tid < 196) ws[tid] = wsq[tid];
  if (tid < 2)   ws[196 + tid] = bsq[tid];
  __syncthreads();
  const int p = blockIdx.x * 256 + tid;
  const int h = p >> 7, w = p & 127;
#pragma unroll
  for (int s = 0; s < 2; ++s) {
    float acc = ws[196 + s];
#pragma unroll
    for (int t = 0; t < 2; ++t) {
      const float* ap = agg + (size_t)(n * 2 + t) * kHW;
      const float* wp = ws + (s * 2 + t) * 49;
      for (int ky = 0; ky < 7; ++ky) {
        int yy = h + ky - 3;
        if ((unsigned)yy < (unsigned)kH)
          for (int kx = 0; kx < 7; ++kx) {
            int xx = w + kx - 3;
            if ((unsigned)xx < (unsigned)kW)
              acc += wp[ky * 7 + kx] * ap[yy * kW + xx];
          }
      }
    }
    sig[(size_t)(n * 2 + s) * kHW + p] = 1.f / (1.f + __expf(-acc));
  }
}

// ---------------------------------------------------------------------------
// Kernel 6: comb = a1c*sig0 + a2c*sig1 (bf16)
// ---------------------------------------------------------------------------
__global__ void combine_kernel(const u16* __restrict__ a1c, const u16* __restrict__ a2c,
                               const float* __restrict__ sig, u16* __restrict__ comb) {
  const int nc = blockIdx.y;                    // n*128 + ch
  const int n = nc >> 7;
  const int p = blockIdx.x * 256 + threadIdx.x;
  const float s0 = sig[(size_t)(n * 2) * kHW + p];
  const float s1 = sig[(size_t)(n * 2 + 1) * kHW + p];
  const size_t o = (size_t)nc * kHW + p;
  comb[o] = f2bf(bf2f(a1c[o]) * s0 + bf2f(a2c[o]) * s1);
}

// ---------------------------------------------------------------------------
// Kernel 7: fused 1x1 conv 128->256 (WMMA) * x, reduced over the 32-px tile
// -> deterministic per-tile partials part[tile][256].  The x-tile (256x32 f32,
// 32KB) is DMA'd into LDS via async loads, overlapped behind the 16 WMMAs.
// ---------------------------------------------------------------------------
__global__ void __launch_bounds__(256)
convfull_reduce_kernel(const u16* __restrict__ comb, const u16* __restrict__ Apk,
                       const float* __restrict__ bc, const float* __restrict__ x,
                       float* __restrict__ part) {
  constexpr int KD = 128, KDP = KD + 8;
  __shared__ u16 Bsh[32 * KDP];
#if HAS_ASYNC_LDS
  __shared__ float Xsh[kC * 32];
#endif
  const int tile = blockIdx.x;
  const int n  = tile >> 9;
  const int p0 = (tile & 511) << 5;
  const int tid = threadIdx.x;
#if HAS_ASYNC_LDS
  {  // per-lane async DMA: thread t owns channel t's 32 px (8x b128)
    const float* gsrc = x + (size_t)(n * kC + tid) * kHW + p0;
    float* ldst = &Xsh[tid * 32];
#pragma unroll
    for (int q = 0; q < 8; ++q) async_cp16(gsrc + q * 4, ldst + q * 4);
  }
#endif
  {  // stage B: 128 channels x 32 px; thread = (channel, half-row of 16 px)
    const int ci = tid & 127, half = tid >> 7;
    const u16* src = comb + (size_t)(n * kCH + ci) * kHW + p0 + half * 16;
    v4u d0 = *(const v4u*)src;
    v4u d1 = *(const v4u*)(src + 8);
    const int c0 = half * 16;
#pragma unroll
    for (int q = 0; q < 4; ++q) {
      Bsh[(c0 + 2 * q + 0) * KDP + ci] = (u16)(d0[q] & 0xffffu);
      Bsh[(c0 + 2 * q + 1) * KDP + ci] = (u16)(d0[q] >> 16);
      Bsh[(c0 + 2 * q + 8) * KDP + ci] = (u16)(d1[q] & 0xffffu);
      Bsh[(c0 + 2 * q + 9) * KDP + ci] = (u16)(d1[q] >> 16);
    }
  }
  __syncthreads();
  const int wave = tid >> 5, lane = tid & 31, col = lane & 15;
  const u16* cb0 = Bsh + col * KDP;
  const u16* cb1 = Bsh + (col + 16) * KDP;
  v8f accA[2], accB[2];
  int rb[2];
#pragma unroll
  for (int j = 0; j < 2; ++j) {
    const int mt = wave + (j << 3);             // m-tile 0..15
    rb[j] = (mt << 4) + ((lane >= 16) ? 8 : 0);
#pragma unroll
    for (int r = 0; r < 8; ++r) {
      float bv = bc[rb[j] + r]; accA[j][r] = bv; accB[j][r] = bv;
    }
#pragma unroll
    for (int ks = 0; ks < 4; ++ks) {
      v16bf a = load_a_pk(Apk, mt * 4 + ks, lane);
      accA[j] = wmma_bf16(a, load_b_lds(cb0, ks * 32, lane), accA[j]);
      accB[j] = wmma_bf16(a, load_b_lds(cb1, ks * 32, lane), accB[j]);
    }
  }
#if HAS_ASYNC_LDS
  wait_async_zero();                            // my DMAs landed in LDS
#endif
  __syncthreads();                              // everyone's DMAs visible
#pragma unroll
  for (int j = 0; j < 2; ++j) {
#pragma unroll
    for (int r = 0; r < 8; ++r) {
      const int c = rb[j] + r;
#if HAS_ASYNC_LDS
      const float x0 = Xsh[c * 32 + col];
      const float x1 = Xsh[c * 32 + col + 16];
#else
      const float* xr = x + (size_t)(n * kC + c) * kHW + p0 + col;
      const float x0 = xr[0];
      const float x1 = xr[16];
#endif
      float v = accA[j][r] * x0 + accB[j][r] * x1;
      v += __shfl_xor(v, 8, 16);
      v += __shfl_xor(v, 4, 16);
      v += __shfl_xor(v, 2, 16);
      v += __shfl_xor(v, 1, 16);
      if (col == 0) part[(size_t)tile * kC + c] = v;
    }
  }
}

// Kernel 8: deterministic partial reduction -> csum (N,256)
__global__ void part_reduce_kernel(const float* __restrict__ part,
                                   float* __restrict__ csum) {
  const int i = blockIdx.x * 256 + threadIdx.x;   // n*256 + c
  const int n = i >> 8, c = i & 255;
  const float* p = part + (size_t)n * kT32PerImg * kC + c;
  float s = 0.f;
  for (int t = 0; t < kT32PerImg; ++t) s += p[(size_t)t * kC];
  csum[i] = s;
}

// ---------------------------------------------------------------------------
// Kernel 9: per-batch sigmoid(mean) -> top-128 of 256 (bitonic in LDS),
// sorted ascending; emits idx (N,128) and inverse map (N,256).
// ---------------------------------------------------------------------------
__global__ void __launch_bounds__(256)
topk_kernel(const float* __restrict__ csum, int* __restrict__ idx,
            int* __restrict__ inv) {
  __shared__ float vals[256];
  __shared__ int   inds[256];
  __shared__ int   sel[128];
  const int n = blockIdx.x, tid = threadIdx.x;
  vals[tid] = 1.f / (1.f + __expf(-csum[n * kC + tid] * (1.f / (float)kHW)));
  inds[tid] = tid;
  __syncthreads();
  for (int ksz = 2; ksz <= 256; ksz <<= 1) {       // descending by value
    for (int j = ksz >> 1; j > 0; j >>= 1) {
      int ixj = tid ^ j;
      if (ixj > tid) {
        bool descending = ((tid & ksz) == 0);
        float v0 = vals[tid], v1 = vals[ixj];
        int   i0 = inds[tid], i1 = inds[ixj];
        bool worse = (v0 < v1) || (v0 == v1 && i0 > i1);
        if (descending ? worse : !worse) {
          vals[tid] = v1; vals[ixj] = v0;
          inds[tid] = i1; inds[ixj] = i0;
        }
      }
      __syncthreads();
    }
  }
  if (tid < 128) sel[tid] = inds[tid];
  __syncthreads();
  for (int ksz = 2; ksz <= 128; ksz <<= 1) {       // selected ids ascending
    for (int j = ksz >> 1; j > 0; j >>= 1) {
      if (tid < 128) {
        int ixj = tid ^ j;
        if (ixj > tid) {
          bool up = ((tid & ksz) == 0);
          int a = sel[tid], b = sel[ixj];
          if (up ? (a > b) : (a < b)) { sel[tid] = b; sel[ixj] = a; }
        }
      }
      __syncthreads();
    }
  }
  inv[n * kC + tid] = -1;
  __syncthreads();
  if (tid < 128) {
    idx[n * kCH + tid] = sel[tid];
    inv[n * kC + sel[tid]] = tid;
  }
}

// ---------------------------------------------------------------------------
// Kernel 10: per-pixel MLP on gathered channels, fully in WMMA.
//   stage1: H = relu(Wf1[64x128] @ E + b1)   (4 waves, 4 K-steps)
//   stage2: O = Wf2[128x64] @ H + b2         (2 M-tiles/wave, 2 K-steps)
// ---------------------------------------------------------------------------
__global__ void __launch_bounds__(128)
mlp_kernel(const float* __restrict__ x, const int* __restrict__ idx,
           const u16* __restrict__ pk1, const float* __restrict__ bf1,
           const u16* __restrict__ pk2, const float* __restrict__ bf2v,
           float* __restrict__ emlp) {
  constexpr int KDP = 136;     // 128 gathered channels, 16B-aligned cols
  constexpr int HDP = 72;      // 64 hidden
  __shared__ u16 Esh[16 * KDP];
  __shared__ u16 Hsh[16 * HDP];
  __shared__ int sidx[128];
  const int tile = blockIdx.x;
  const int n  = tile >> 10;
  const int p0 = (tile & 1023) << 4;
  const int tid = threadIdx.x;
  sidx[tid] = idx[n * kCH + tid];
  __syncthreads();
  {  // gather one channel row (16 px) per thread, f32 -> bf16, col-major LDS
    const int c = sidx[tid];
    const float4* xp = (const float4*)(x + (size_t)(n * kC + c) * kHW + p0);
#pragma unroll
    for (int q = 0; q < 4; ++q) {
      float4 v = xp[q];
      Esh[(q * 4 + 0) * KDP + tid] = f2bf(v.x);
      Esh[(q * 4 + 1) * KDP + tid] = f2bf(v.y);
      Esh[(q * 4 + 2) * KDP + tid] = f2bf(v.z);
      Esh[(q * 4 + 3) * KDP + tid] = f2bf(v.w);
    }
  }
  __syncthreads();
  const int wave = tid >> 5, lane = tid & 31, col = lane & 15;
  {  // ---- stage 1: hidden (64 x 16) ----
    const int rbase = (wave << 4) + ((lane >= 16) ? 8 : 0);
    v8f acc;
#pragma unroll
    for (int r = 0; r < 8; ++r) acc[r] = bf1[rbase + r];
    const u16* cb = Esh + col * KDP;
#pragma unroll
    for (int ks = 0; ks < 4; ++ks) {
      v16bf a = load_a_pk(pk1, wave * 4 + ks, lane);
      acc = wmma_bf16(a, load_b_lds(cb, ks * 32, lane), acc);
    }
#pragma unroll
    for (int r = 0; r < 8; ++r)
      Hsh[col * HDP + rbase + r] = f2bf(fmaxf(acc[r], 0.f));
  }
  __syncthreads();
#pragma unroll
  for (int j = 0; j < 2; ++j) {  // ---- stage 2: out (128 x 16) ----
    const int mt = (wave << 1) + j;
    const int rbase = (mt << 4) + ((lane >= 16) ? 8 : 0);
    v8f acc;
#pragma unroll
    for (int r = 0; r < 8; ++r) acc[r] = bf2v[rbase + r];
    const u16* cb = Hsh + col * HDP;
#pragma unroll
    for (int ks = 0; ks < 2; ++ks) {
      v16bf a = load_a_pk(pk2, mt * 2 + ks, lane);
      acc = wmma_bf16(a, load_b_lds(cb, ks * 32, lane), acc);
    }
    const size_t ob = (size_t)(n * kCH + rbase) * kHW + p0 + col;
#pragma unroll
    for (int r = 0; r < 8; ++r) emlp[ob + (size_t)r * kHW] = acc[r];
  }
}

// ---------------------------------------------------------------------------
// Kernel 11: out = x with selected channels replaced by the OTHER image's MLP
// ---------------------------------------------------------------------------
__global__ void scatter_kernel(const float* __restrict__ x,
                               const float* __restrict__ emlp_other,
                               const int* __restrict__ inv,
                               float* __restrict__ out) {
  const int nc = blockIdx.y;                    // n*256 + c
  const int n = nc >> 8;
  const int p = blockIdx.x * 256 + threadIdx.x;
  const int slot = inv[nc];
  const size_t o = (size_t)nc * kHW + p;
  out[o] = (slot >= 0) ? emlp_other[(size_t)(n * kCH + slot) * kHW + p] : x[o];
}

// ---------------------------------------------------------------------------
extern "C" void kernel_launch(void* const* d_in, const int* in_sizes, int n_in,
                              void* d_out, int out_size, void* d_ws, size_t ws_size,
                              hipStream_t stream) {
  (void)in_sizes; (void)n_in; (void)out_size; (void)ws_size;
  const float* x1        = (const float*)d_in[0];
  const float* x2        = (const float*)d_in[1];
  const float* w_conv0   = (const float*)d_in[2];
  const float* b_conv0   = (const float*)d_in[3];
  const float* w_spatial = (const float*)d_in[4];
  const float* b_spatial = (const float*)d_in[5];
  const float* w_conv1   = (const float*)d_in[6];
  const float* b_conv1   = (const float*)d_in[7];
  const float* w_conv2   = (const float*)d_in[8];
  const float* b_conv2   = (const float*)d_in[9];
  const float* w_squeeze = (const float*)d_in[10];
  const float* b_squeeze = (const float*)d_in[11];
  const float* w_conv    = (const float*)d_in[12];
  const float* b_conv    = (const float*)d_in[13];
  const float* w_fc1     = (const float*)d_in[14];
  const float* b_fc1     = (const float*)d_in[15];
  const float* w_fc2     = (const float*)d_in[16];
  const float* b_fc2     = (const float*)d_in[17];

  char* ws = (char*)d_ws;
  size_t off = 0;
  auto alloc = [&](size_t bytes) -> void* {
    void* p = ws + off;
    off += (bytes + 255) & ~(size_t)255;
    return p;
  };
  u16*   a1    = (u16*)  alloc(kNCHW * 2);
  u16*   a2    = (u16*)  alloc(kNCHW * 2);
  u16*   a1c   = (u16*)  alloc(kNKHW * 2);
  u16*   a2c   = (u16*)  alloc(kNKHW * 2);
  float* agg   = (float*)alloc((size_t)kN * 2 * kHW * 4);
  float* sig   = (float*)alloc((size_t)kN * 2 * kHW * 4);
  u16*   comb  = (u16*)  alloc(kNKHW * 2);
  float* part  = (float*)alloc((size_t)kT32 * kC * 4);
  float* csum  = (float*)alloc((size_t)kN * kC * 4);
  int*   idx1  = (int*)  alloc((size_t)kN * kCH * 4);
  int*   idx2  = (int*)  alloc((size_t)kN * kCH * 4);
  int*   inv1  = (int*)  alloc((size_t)kN * kC * 4);
  int*   inv2  = (int*)  alloc((size_t)kN * kC * 4);
  float* emlp1 = (float*)alloc(kNKHW * 4);
  float* emlp2 = (float*)alloc(kNKHW * 4);
  // pre-packed bf16 A-fragment weights
  u16* pk_c1   = (u16*)  alloc((size_t)kCH * kC * 2);
  u16* pk_c2   = (u16*)  alloc((size_t)kCH * kC * 2);
  u16* pk_full = (u16*)  alloc((size_t)kC * kCH * 2);
  u16* pk_fc1  = (u16*)  alloc((size_t)kHID * kCH * 2);
  u16* pk_fc2  = (u16*)  alloc((size_t)kCH * kHID * 2);

  float* out1 = (float*)d_out;
  float* out2 = out1 + kNCHW;

  // weight packing (deterministic, tiny)
  pack_a_kernel<<<(kCH * kC) / 256, 256, 0, stream>>>(w_conv1, pk_c1, kCH, kC);
  pack_a_kernel<<<(kCH * kC) / 256, 256, 0, stream>>>(w_conv2, pk_c2, kCH, kC);
  pack_a_kernel<<<(kC * kCH) / 256, 256, 0, stream>>>(w_conv, pk_full, kC, kCH);
  pack_a_kernel<<<(kHID * kCH) / 256, 256, 0, stream>>>(w_fc1, pk_fc1, kHID, kCH);
  pack_a_kernel<<<(kCH * kHID) / 256, 256, 0, stream>>>(w_fc2, pk_fc2, kCH, kHID);

  const dim3 gridNC(kHW / 256, kN * kC);
  const dim3 gridNK(kHW / 256, kN * kCH);
  const dim3 gridN(kHW / 256, kN);

  for (int t = 0; t < 2; ++t) {
    const float* x   = t ? x2 : x1;
    int* idx         = t ? idx2 : idx1;
    int* inv         = t ? inv2 : inv1;
    float* emlp      = t ? emlp2 : emlp1;

    dw5_kernel<<<gridNC, 256, 0, stream>>>(x, w_conv0, b_conv0, a1);
    dw7d3_kernel<<<gridNC, 256, 0, stream>>>(a1, w_spatial, b_spatial, a2);
    conv1x1_half_kernel<<<kT32, 256, 0, stream>>>(a1, pk_c1, b_conv1, a1c);
    conv1x1_half_kernel<<<kT32, 256, 0, stream>>>(a2, pk_c2, b_conv2, a2c);
    agg_kernel<<<gridN, 256, 0, stream>>>(a1c, a2c, agg);
    squeeze_kernel<<<gridN, 256, 0, stream>>>(agg, w_squeeze, b_squeeze, sig);
    combine_kernel<<<gridNK, 256, 0, stream>>>(a1c, a2c, sig, comb);
    convfull_reduce_kernel<<<kT32, 256, 0, stream>>>(comb, pk_full, b_conv, x, part);
    part_reduce_kernel<<<(kN * kC) / 256, 256, 0, stream>>>(part, csum);
    topk_kernel<<<kN, 256, 0, stream>>>(csum, idx, inv);
    mlp_kernel<<<kT16, 128, 0, stream>>>(x, idx, pk_fc1, b_fc1, pk_fc2, b_fc2, emlp);
  }
  // cross-exchange: x1 slots get MLP(e2), x2 slots get MLP(e1)
  scatter_kernel<<<gridNC, 256, 0, stream>>>(x1, emlp2, inv1, out1);
  scatter_kernel<<<gridNC, 256, 0, stream>>>(x2, emlp1, inv2, out2);
}
